// RWKV_49057116455669
// MI455X (gfx1250) — matrix operand order repaired and checked by reading, not compiled, not run on hardware
//
#include <hip/hip_runtime.h>

// ---------------------------------------------------------------------------
// RWKV forward for MI455X (gfx1250, wave32, WMMA).
// GEMMs: bf16 A/B, f32 accum via v_wmma_f32_16x16x32_bf16.
// A tiles staged into LDS with GLOBAL_LOAD_ASYNC_TO_LDS_B128 (ASYNCcnt),
// B tiles transposed through VGPRs into LDS, next B tile prefetched.
// Weights converted f32->bf16 into d_ws each call (deterministic).
// Workspace requirement ~340 MB.
// ---------------------------------------------------------------------------

typedef __bf16 bh16;
typedef __attribute__((ext_vector_type(16))) __bf16 v16bf;
typedef __attribute__((ext_vector_type(8)))  float  v8f;
typedef __attribute__((ext_vector_type(4)))  int    rwkv_v4i;

constexpr int L_SEQ = 4096;
constexpr int EDIM  = 1024;
constexpr int NLAYER = 8;
constexpr int HDIM  = 4096;

#if defined(__AMDGCN__) && __has_builtin(__builtin_amdgcn_global_load_async_to_lds_b128)
#define RWKV_HAS_ASYNC 1
#else
#define RWKV_HAS_ASYNC 0
#endif

// Async 16-byte global->LDS copy (per-lane), fallback = synchronous copy.
__device__ __forceinline__ void rwkv_async_cp16(const bh16* g, bh16* l) {
#if RWKV_HAS_ASYNC
    __builtin_amdgcn_global_load_async_to_lds_b128(
        (__attribute__((address_space(1))) rwkv_v4i*)g,
        (__attribute__((address_space(3))) rwkv_v4i*)l, 0, 0);
#else
    *(uint4*)l = *(const uint4*)g;
#endif
}

__device__ __forceinline__ void rwkv_wait_async() {
#if RWKV_HAS_ASYNC
#if __has_builtin(__builtin_amdgcn_s_wait_asynccnt)
    __builtin_amdgcn_s_wait_asynccnt(0);
#else
    asm volatile("s_wait_asynccnt 0x0" ::: "memory");
#endif
#endif
}

// ------------------------------- helpers -----------------------------------

__global__ void rwkv_f32_to_bf16(const float* __restrict__ src,
                                 bh16* __restrict__ dst, size_t n) {
    size_t i = (size_t)blockIdx.x * blockDim.x + threadIdx.x;
    size_t stride = (size_t)gridDim.x * blockDim.x;
    for (; i < n; i += stride) dst[i] = (bh16)src[i];
}

__global__ void rwkv_embed(const int* __restrict__ s,
                           const float* __restrict__ embed,
                           float* __restrict__ x) {
    int i = blockIdx.x * blockDim.x + threadIdx.x;
    if (i >= L_SEQ * EDIM) return;
    int t = i / EDIM, e = i % EDIM;
    int tok = (t == 0) ? 0 : s[t - 1];
    x[i] = embed[tok * EDIM + e];
}

__global__ void rwkv_tm_pre(const float* __restrict__ x,
                            const float* __restrict__ mk,
                            const float* __restrict__ mv,
                            const float* __restrict__ mr,
                            bh16* __restrict__ kx, bh16* __restrict__ vx,
                            bh16* __restrict__ rx) {
    int i = blockIdx.x * blockDim.x + threadIdx.x;
    if (i >= L_SEQ * EDIM) return;
    int t = i / EDIM, e = i % EDIM;
    float xv = x[i];
    float sx = (t == 0) ? 0.f : x[i - EDIM];
    float a = mk[e], b = mv[e], c = mr[e];
    kx[i] = (bh16)(xv * a + sx * (1.f - a));
    vx[i] = (bh16)(xv * b + sx * (1.f - b));
    rx[i] = (bh16)(xv * c + sx * (1.f - c));
}

__global__ void rwkv_cm_pre(const float* __restrict__ x,
                            const float* __restrict__ mk,
                            const float* __restrict__ mr,
                            bh16* __restrict__ xk, bh16* __restrict__ xr) {
    int i = blockIdx.x * blockDim.x + threadIdx.x;
    if (i >= L_SEQ * EDIM) return;
    int t = i / EDIM, e = i % EDIM;
    float xv = x[i];
    float sx = (t == 0) ? 0.f : x[i - EDIM];
    float a = mk[e], c = mr[e];
    xk[i] = (bh16)(xv * a + sx * (1.f - a));
    xr[i] = (bh16)(xv * c + sx * (1.f - c));
}

// Per-channel max-stabilized WKV scan fused with sigmoid(r) gate.
__global__ void rwkv_wkv_gate(const float* __restrict__ k,
                              const float* __restrict__ v,
                              const float* __restrict__ rlin,
                              const float* __restrict__ tfirst,
                              const float* __restrict__ tdecay,
                              bh16* __restrict__ outg) {
    int e = blockIdx.x * blockDim.x + threadIdx.x;
    if (e >= EDIM) return;
    float tf = tfirst[e];
    float wd = __expf(tdecay[e]);
    float aa = 0.f, bb = 0.f, pp = -__builtin_inff();
    for (int t = 0; t < L_SEQ; ++t) {
        size_t idx = (size_t)t * EDIM + e;
        float kk = k[idx], vv = v[idx];
        float ww = tf + kk;
        float p  = fmaxf(pp, ww);
        float e1 = __expf(pp - p);
        float e2 = __expf(ww - p);
        float o  = (e1 * aa + e2 * vv) / (e1 * bb + e2);
        float w2 = pp - wd;
        float p2 = fmaxf(w2, kk);
        e1 = __expf(w2 - p2);
        e2 = __expf(kk - p2);
        aa = e1 * aa + e2 * vv;
        bb = e1 * bb + e2;
        pp = p2;
        float r = 1.f / (1.f + __expf(-rlin[idx]));
        outg[idx] = (bh16)(r * o);
    }
}

__global__ void rwkv_zero_out(float* out) { out[0] = 0.f; }

// One wave per token: logits[4] = y[t,:] @ W[E,4]; logsoftmax*0.5; gather s[t].
__global__ void rwkv_head(const float* __restrict__ y,
                          const float* __restrict__ W,
                          const int* __restrict__ s,
                          float* __restrict__ out) {
    int gtid = blockIdx.x * blockDim.x + threadIdx.x;
    int wave = gtid >> 5;
    int lane = gtid & 31;
    if (wave >= L_SEQ) return;
    const float* yr = y + (size_t)wave * EDIM;
    float l0 = 0.f, l1 = 0.f, l2 = 0.f, l3 = 0.f;
    for (int e = lane; e < EDIM; e += 32) {
        float yv = yr[e];
        const float* w = W + e * 4;
        l0 += yv * w[0]; l1 += yv * w[1]; l2 += yv * w[2]; l3 += yv * w[3];
    }
    #pragma unroll
    for (int offd = 16; offd > 0; offd >>= 1) {
        l0 += __shfl_down(l0, offd, 32);
        l1 += __shfl_down(l1, offd, 32);
        l2 += __shfl_down(l2, offd, 32);
        l3 += __shfl_down(l3, offd, 32);
    }
    if (lane == 0) {
        float m = fmaxf(fmaxf(l0, l1), fmaxf(l2, l3));
        float se = __expf(l0 - m) + __expf(l1 - m) + __expf(l2 - m) + __expf(l3 - m);
        float lse = m + __logf(se);
        float lv[4] = {l0, l1, l2, l3};
        float lp = (lv[s[wave]] - lse) * 0.5f;
        atomicAdd(out, lp);
    }
}

// ------------------------------ WMMA GEMM ----------------------------------
// D[M,N] = epilogue(A[M,K] @ B[K,N]); A,B bf16 row-major; accumulate f32.
// Block tile 128x128, 8 waves (wave32), each wave 32x64 = 2x4 WMMA 16x16 tiles.
// MODE 0: Df = acc
// MODE 1: Df += acc                      (residual)
// MODE 2: Dh = bf16(relu(acc)^2)        (channel-mix key)
// MODE 3: Df += sigmoid(G) * acc        (gated residual)
// MODE 4: Df = gelu(acc + bias[n])      (neck)

#define TILE_K    32
#define LDS_PITCH 40   // bf16 elems per LDS row (32 data + 8 pad => 80 B)

union RwkvFrag { uint4 u[2]; v16bf v; };

template <int MODE>
__global__ __launch_bounds__(256)
void rwkv_gemm_bf16(const bh16* __restrict__ A, const bh16* __restrict__ B,
                    float* __restrict__ Df, bh16* __restrict__ Dh,
                    const float* __restrict__ G, const float* __restrict__ bias,
                    int M, int N, int K) {
    __shared__ bh16 sA[128 * LDS_PITCH];
    __shared__ bh16 sB[128 * LDS_PITCH];

    const int tid  = threadIdx.x;
    const int lane = tid & 31;
    const int wave = tid >> 5;
    const int wm   = wave & 3;   // wave row: 4 x 32 = 128 M
    const int wn   = wave >> 2;  // wave col: 2 x 64 = 128 N
    const int bM   = blockIdx.y * 128;
    const int bN   = blockIdx.x * 128;
    const int half = lane >> 4;  // K-half select per ISA fragment layout
    const int l15  = lane & 15;

    v8f acc[2][4];
    #pragma unroll
    for (int i = 0; i < 2; ++i)
        #pragma unroll
        for (int j = 0; j < 4; ++j)
            #pragma unroll
            for (int r = 0; r < 8; ++r) acc[i][j][r] = 0.f;

    for (int k0 = 0; k0 < K; k0 += TILE_K) {
        // --- A tile 128x32 (row-major): async DMA straight into LDS. ---
        #pragma unroll
        for (int u = 0; u < 2; ++u) {
            int c = tid + u * 256;           // 512 chunks of 8 bf16
            int row = c >> 2;
            int col = (c & 3) << 3;
            rwkv_async_cp16(A + (size_t)(bM + row) * K + k0 + col,
                            sA + row * LDS_PITCH + col);
        }
        // --- B tile 32x128: both loads in flight, then scatter-transpose. ---
        uint4 breg[2];
        int bk[2], bn[2];
        #pragma unroll
        for (int u = 0; u < 2; ++u) {
            int c = tid + u * 256;
            bk[u] = c >> 4;                  // k row 0..31
            bn[u] = (c & 15) << 3;           // n chunk of 8
            breg[u] = *(const uint4*)(B + (size_t)(k0 + bk[u]) * N + bN + bn[u]);
        }
        if (k0 + TILE_K < K)                 // hint next B tile toward L2/L0
            __builtin_prefetch(B + (size_t)(k0 + TILE_K + (tid >> 4)) * N
                                   + bN + ((tid & 15) << 3), 0, 0);
        #pragma unroll
        for (int u = 0; u < 2; ++u) {
            union { uint4 u4; bh16 h[8]; } d;
            d.u4 = breg[u];
            #pragma unroll
            for (int j = 0; j < 8; ++j)
                sB[(bn[u] + j) * LDS_PITCH + bk[u]] = d.h[j];
        }
        rwkv_wait_async();                   // A-tile async copies complete
        __syncthreads();

        // A frag (16x32 bf16): lane l15 = row, elements 0..7 -> K=8*half..,
        // elements 8..15 -> K=8*half+16.. (ISA 16-bit A layout).
        RwkvFrag a[2], b[4];
        #pragma unroll
        for (int i = 0; i < 2; ++i) {
            const bh16* p = sA + (wm * 32 + i * 16 + l15) * LDS_PITCH + half * 8;
            a[i].u[0] = *(const uint4*)(p);
            a[i].u[1] = *(const uint4*)(p + 16);
        }
        // B frag (32x16 bf16): lane l15 = col n, elements 0..7 -> K=8*half..,
        // elements 8..15 -> K=8*half+16.. (same K order as A).
        #pragma unroll
        for (int j = 0; j < 4; ++j) {
            const bh16* p = sB + (wn * 64 + j * 16 + l15) * LDS_PITCH + half * 8;
            b[j].u[0] = *(const uint4*)(p);
            b[j].u[1] = *(const uint4*)(p + 16);
        }

        #pragma unroll
        for (int i = 0; i < 2; ++i)
            #pragma unroll
            for (int j = 0; j < 4; ++j)
                acc[i][j] = __builtin_amdgcn_wmma_f32_16x16x32_bf16(
                    false, a[i].v, false, b[j].v, (short)0, acc[i][j],
                    false, false);
        __syncthreads();
    }

    // Epilogue; C/D layout: VGPR r -> M = r + 8*half, lane l15 -> N.
    #pragma unroll
    for (int i = 0; i < 2; ++i) {
        #pragma unroll
        for (int j = 0; j < 4; ++j) {
            int n  = bN + wn * 64 + j * 16 + l15;
            int mb = bM + wm * 32 + i * 16 + (half << 3);
            #pragma unroll
            for (int r = 0; r < 8; ++r) {
                int m = mb + r;
                float v = acc[i][j][r];
                size_t idx = (size_t)m * N + n;
                if constexpr (MODE == 0) {
                    Df[idx] = v;
                } else if constexpr (MODE == 1) {
                    Df[idx] += v;
                } else if constexpr (MODE == 2) {
                    float t = v > 0.f ? v : 0.f;
                    Dh[idx] = (bh16)(t * t);
                } else if constexpr (MODE == 3) {
                    float g = 1.f / (1.f + __expf(-G[idx]));
                    Df[idx] += g * v;
                } else if constexpr (MODE == 4) {
                    float t = v + bias[n];
                    float c = 0.7978845608028654f * (t + 0.044715f * t * t * t);
                    Df[idx] = 0.5f * t * (1.f + tanhf(c));
                }
            }
        }
    }
}

// ------------------------------- driver ------------------------------------

extern "C" void kernel_launch(void* const* d_in, const int* in_sizes, int n_in,
                              void* d_out, int out_size, void* d_ws, size_t ws_size,
                              hipStream_t stream) {
    const int*   s        = (const int*)  d_in[0];
    const float* embed    = (const float*)d_in[1];
    const float* tm_first = (const float*)d_in[2];
    const float* tm_decay = (const float*)d_in[3];
    const float* tm_mk    = (const float*)d_in[4];
    const float* tm_mv    = (const float*)d_in[5];
    const float* tm_mr    = (const float*)d_in[6];
    const float* tm_Wk    = (const float*)d_in[7];
    const float* tm_Wv    = (const float*)d_in[8];
    const float* tm_Wr    = (const float*)d_in[9];
    const float* tm_Wo    = (const float*)d_in[10];
    const float* cm_mk    = (const float*)d_in[11];
    const float* cm_mr    = (const float*)d_in[12];
    const float* cm_Wk    = (const float*)d_in[13];
    const float* cm_Wr    = (const float*)d_in[14];
    const float* cm_Wv    = (const float*)d_in[15];
    const float* neck_W   = (const float*)d_in[16];
    const float* neck_b   = (const float*)d_in[17];
    const float* head_W   = (const float*)d_in[18];
    float* out = (float*)d_out;

    const size_t EE = (size_t)EDIM * EDIM;
    const size_t EH = (size_t)EDIM * HDIM;
    const size_t LE = (size_t)L_SEQ * EDIM;
    const size_t LH = (size_t)L_SEQ * HDIM;

    size_t off = 0;
    char* base = (char*)d_ws;
    auto alloc = [&](size_t bytes) -> void* {
        off = (off + 255) & ~(size_t)255;
        void* r = base + off;
        off += bytes;
        return r;
    };

    // bf16 weights
    bh16* wtmWk = (bh16*)alloc(NLAYER * EE * 2);
    bh16* wtmWv = (bh16*)alloc(NLAYER * EE * 2);
    bh16* wtmWr = (bh16*)alloc(NLAYER * EE * 2);
    bh16* wtmWo = (bh16*)alloc(NLAYER * EE * 2);
    bh16* wcmWk = (bh16*)alloc(NLAYER * EH * 2);
    bh16* wcmWr = (bh16*)alloc(NLAYER * EE * 2);
    bh16* wcmWv = (bh16*)alloc(NLAYER * EH * 2);
    bh16* wneck = (bh16*)alloc(EE * 2);
    // activations
    float* x   = (float*)alloc(LE * 4);
    bh16*  kx  = (bh16*) alloc(LE * 2);   // also xk / x_bf16 for neck
    bh16*  vx  = (bh16*) alloc(LE * 2);   // also r*wkv gate output
    bh16*  rx  = (bh16*) alloc(LE * 2);   // also xr
    float* kf  = (float*)alloc(LE * 4);   // k projections; also neck output y
    float* vf  = (float*)alloc(LE * 4);
    float* rl  = (float*)alloc(LE * 4);   // r pre-sigmoid
    bh16*  kkb = (bh16*) alloc(LH * 2);   // relu(xk@Wk)^2, bf16
    (void)ws_size; (void)in_sizes; (void)n_in; (void)out_size;

    // --- weight conversion f32 -> bf16 ---
    auto conv = [&](const float* src, bh16* dst, size_t n) {
        rwkv_f32_to_bf16<<<2048, 256, 0, stream>>>(src, dst, n);
    };
    conv(tm_Wk, wtmWk, NLAYER * EE);
    conv(tm_Wv, wtmWv, NLAYER * EE);
    conv(tm_Wr, wtmWr, NLAYER * EE);
    conv(tm_Wo, wtmWo, NLAYER * EE);
    conv(cm_Wk, wcmWk, NLAYER * EH);
    conv(cm_Wr, wcmWr, NLAYER * EE);
    conv(cm_Wv, wcmWv, NLAYER * EH);
    conv(neck_W, wneck, EE);

    // --- embed ---
    int gLE = (int)((LE + 255) / 256);
    rwkv_embed<<<gLE, 256, 0, stream>>>(s, embed, x);

    dim3 gE(EDIM / 128, L_SEQ / 128);   // N = 1024
    dim3 gH(HDIM / 128, L_SEQ / 128);   // N = 4096

    for (int l = 0; l < NLAYER; ++l) {
        size_t oE  = (size_t)l * EDIM;
        size_t oEE = (size_t)l * EE;
        size_t oEH = (size_t)l * EH;

        // Time mixing
        rwkv_tm_pre<<<gLE, 256, 0, stream>>>(x, tm_mk + oE, tm_mv + oE,
                                             tm_mr + oE, kx, vx, rx);
        rwkv_gemm_bf16<0><<<gE, 256, 0, stream>>>(kx, wtmWk + oEE, kf, nullptr,
                                                  nullptr, nullptr, L_SEQ, EDIM, EDIM);
        rwkv_gemm_bf16<0><<<gE, 256, 0, stream>>>(vx, wtmWv + oEE, vf, nullptr,
                                                  nullptr, nullptr, L_SEQ, EDIM, EDIM);
        rwkv_gemm_bf16<0><<<gE, 256, 0, stream>>>(rx, wtmWr + oEE, rl, nullptr,
                                                  nullptr, nullptr, L_SEQ, EDIM, EDIM);
        rwkv_wkv_gate<<<EDIM / 256, 256, 0, stream>>>(kf, vf, rl, tm_first + oE,
                                                      tm_decay + oE, vx);
        rwkv_gemm_bf16<1><<<gE, 256, 0, stream>>>(vx, wtmWo + oEE, x, nullptr,
                                                  nullptr, nullptr, L_SEQ, EDIM, EDIM);

        // Channel mixing
        rwkv_cm_pre<<<gLE, 256, 0, stream>>>(x, cm_mk + oE, cm_mr + oE, kx, rx);
        rwkv_gemm_bf16<2><<<gH, 256, 0, stream>>>(kx, wcmWk + oEH, nullptr, kkb,
                                                  nullptr, nullptr, L_SEQ, HDIM, EDIM);
        rwkv_gemm_bf16<0><<<gE, 256, 0, stream>>>(rx, wcmWr + oEE, rl, nullptr,
                                                  nullptr, nullptr, L_SEQ, EDIM, EDIM);
        rwkv_gemm_bf16<3><<<gE, 256, 0, stream>>>(kkb, wcmWv + oEH, x, nullptr,
                                                  rl, nullptr, L_SEQ, EDIM, HDIM);
    }

    // Neck: y = gelu(x @ neck_W + b)
    rwkv_f32_to_bf16<<<2048, 256, 0, stream>>>(x, kx, LE);
    rwkv_gemm_bf16<4><<<gE, 256, 0, stream>>>(kx, wneck, kf, nullptr, nullptr,
                                              neck_b, L_SEQ, EDIM, EDIM);

    // Head + logsoftmax + gather + sum
    rwkv_zero_out<<<1, 1, 0, stream>>>(out);
    rwkv_head<<<(L_SEQ * 32) / 256, 256, 0, stream>>>(kf, head_W, s, out);
}
